// SlotAttentionModule_50036368999068
// MI455X (gfx1250) — matrix operand order repaired and checked by reading, not compiled, not run on hardware
//
#include <hip/hip_runtime.h>
#include <hip/hip_bf16.h>

typedef __bf16 bf16_t;
typedef __attribute__((ext_vector_type(16))) __bf16 v16bf;
typedef __attribute__((ext_vector_type(8)))  float  v8f;
typedef __attribute__((ext_vector_type(4)))  int    v4i;

#if defined(__AMDGCN__) && __has_builtin(__builtin_amdgcn_global_load_async_to_lds_b128) && \
    __has_builtin(__builtin_amdgcn_s_wait_asynccnt)
#define USE_ASYNC_LDS 1
#else
#define USE_ASYNC_LDS 0
#endif

union Frag16 {
    v16bf v;
    uint4 q[2];
    bf16_t h[16];
};

// ---- WMMA helpers (CDNA5 layouts per CDNA5 ISA) -----------------

__device__ __forceinline__ v8f wmma_bf16(v16bf a, v16bf b, v8f c) {
    // (neg_a, A, neg_b, B, c_mod, C, reuse_a, reuse_b)
    return __builtin_amdgcn_wmma_f32_16x16x32_bf16(false, a, false, b, (short)0, c,
                                                   false, false);
}

// A-matrix 16x32 bf16 fragment. Caller passes &buf[row][k_off] for row = lane&15.
// Lanes 0-15 hold K=0..15, lanes 16-31 hold K=16..31.
__device__ __forceinline__ v16bf frag_a_16x32(const bf16_t* row_ptr) {
    const int lane = threadIdx.x & 31;
    const int base = (lane < 16) ? 0 : 16;
    Frag16 f;
    f.q[0] = *(const uint4*)(row_ptr + base);
    f.q[1] = *(const uint4*)(row_ptr + base + 8);
    return f.v;
}

// B-matrix 32x16 bf16 fragment, K contiguous in memory per column.
// Lanes 0-15 hold K=0..15, lanes 16-31 hold K=16..31.
__device__ __forceinline__ v16bf frag_b_contig(const bf16_t* col_ptr) {
    const int lane = threadIdx.x & 31;
    const int off = (lane < 16) ? 0 : 16;
    Frag16 f;
    f.q[0] = *(const uint4*)(col_ptr + off);
    f.q[1] = *(const uint4*)(col_ptr + off + 8);
    return f.v;
}

// B-matrix 32x16 bf16 fragment, K strided (row-major [K][stride] source).
__device__ __forceinline__ v16bf frag_b_strided(const bf16_t* base_ptr, int stride) {
    const int lane = threadIdx.x & 31;
    const int k0 = (lane < 16) ? 0 : 16;
    Frag16 f;
#pragma unroll
    for (int j = 0; j < 16; ++j) f.h[j] = base_ptr[(k0 + j) * stride];
    return f.v;
}

// Stage one 32x64 bf16 chunk pair (k,v) into this wave's LDS buffers.
// Async path: GLOBAL_LOAD_ASYNC_TO_LDS_B128, tracked by ASYNCcnt (no VGPR round-trip).
__device__ __forceinline__ void stage_chunk(bf16_t* dstk, bf16_t* dstv,
                                            const bf16_t* srck, const bf16_t* srcv,
                                            int lane) {
#if USE_ASYNC_LDS
#pragma unroll
    for (int t = 0; t < 8; ++t) {
        const int off = (lane + 32 * t) * 8;  // bf16 elems; 16B per b128
        __builtin_amdgcn_global_load_async_to_lds_b128(
            (__attribute__((address_space(1))) v4i*)(srck + off),
            (__attribute__((address_space(3))) v4i*)(dstk + off), 0, 0);
        __builtin_amdgcn_global_load_async_to_lds_b128(
            (__attribute__((address_space(1))) v4i*)(srcv + off),
            (__attribute__((address_space(3))) v4i*)(dstv + off), 0, 0);
    }
#else
    __builtin_prefetch(srck, 0, 0);
    __builtin_prefetch(srcv, 0, 0);
#pragma unroll
    for (int t = 0; t < 8; ++t) {
        ((uint4*)dstk)[lane + 32 * t] = ((const uint4*)srck)[lane + 32 * t];
        ((uint4*)dstv)[lane + 32 * t] = ((const uint4*)srcv)[lane + 32 * t];
    }
#endif
}

__device__ __forceinline__ void wait_chunk(bool more_inflight) {
#if USE_ASYNC_LDS
    if (more_inflight) __builtin_amdgcn_s_wait_asynccnt(16);
    else               __builtin_amdgcn_s_wait_asynccnt(0);
#endif
}

__device__ __forceinline__ float sigm(float x) { return 1.0f / (1.0f + __expf(-x)); }

// ---- Kernel 0: convert projection weights to bf16 --------------------------

__global__ void convert_weights(const float* __restrict__ Wk,
                                const float* __restrict__ Wv,
                                bf16_t* __restrict__ wkb,
                                bf16_t* __restrict__ wvb) {
    int i = blockIdx.x * blockDim.x + threadIdx.x;
    if (i < 64 * 256) {
        wkb[i] = (bf16_t)Wk[i];
        wvb[i] = (bf16_t)Wv[i];
    }
}

// ---- Kernel 1: fused feature LayerNorm + K/V projection (bf16 WMMA) --------
// One wave per 16 feature rows. k[n,h] = sum_d LN(f)[n,d] * Wk[h,d].

__global__ __launch_bounds__(32) void ln_project(
    const float* __restrict__ feats,
    const float* __restrict__ lng, const float* __restrict__ lnb,
    const bf16_t* __restrict__ wkb, const bf16_t* __restrict__ wvb,
    bf16_t* __restrict__ kb, bf16_t* __restrict__ vb)
{
    __shared__ bf16_t s_lnf[16][256];   // 8 KB
    const int lane = threadIdx.x;
    const long row0 = (long)blockIdx.x * 16;

    for (int r = 0; r < 16; ++r) {
        const float* fr = feats + (row0 + r) * 256;
        float vals[8];
        float s = 0.f;
#pragma unroll
        for (int j = 0; j < 8; ++j) { vals[j] = fr[lane + 32 * j]; s += vals[j]; }
        for (int m = 16; m > 0; m >>= 1) s += __shfl_xor(s, m, 32);
        const float mean = s * (1.0f / 256.0f);
        float vs = 0.f;
#pragma unroll
        for (int j = 0; j < 8; ++j) { float d = vals[j] - mean; vs += d * d; }
        for (int m = 16; m > 0; m >>= 1) vs += __shfl_xor(vs, m, 32);
        const float rstd = rsqrtf(vs * (1.0f / 256.0f) + 1e-5f);
#pragma unroll
        for (int j = 0; j < 8; ++j) {
            const int c = lane + 32 * j;
            s_lnf[r][c] = (bf16_t)((vals[j] - mean) * rstd * lng[c] + lnb[c]);
        }
    }
    __syncthreads();

    v16bf afr[8];
#pragma unroll
    for (int t = 0; t < 8; ++t)
        afr[t] = frag_a_16x32(&s_lnf[lane & 15][t * 32]);

    for (int mat = 0; mat < 2; ++mat) {
        const bf16_t* W = mat ? wvb : wkb;
        bf16_t* out = mat ? vb : kb;
#pragma unroll
        for (int g = 0; g < 4; ++g) {
            const int h = g * 16 + (lane & 15);
            v8f c = {0.f, 0.f, 0.f, 0.f, 0.f, 0.f, 0.f, 0.f};
#pragma unroll
            for (int t = 0; t < 8; ++t)
                c = wmma_bf16(afr[t], frag_b_contig(W + h * 256 + t * 32), c);
            const long rbase = row0 + ((lane < 16) ? 0 : 8);
#pragma unroll
            for (int r = 0; r < 8; ++r)
                out[(rbase + r) * 64 + h] = (bf16_t)c[r];
        }
    }
}

// ---- Kernel 2: fused slot-attention iterations -----------------------------
// 8 waves per batch; each wave streams a disjoint 512-point slice of N through
// its own double-buffered LDS staging buffers (async LDS copies), accumulating
// un-normalized updates U[6][64] and Z[6] in WMMA accumulators. Cross-wave
// merge via ds_add_f32 atomics. Softmax over the slot axis is local per point;
// the /(sum_n + eps) normalization commutes with attn@v, so no attention
// matrix is ever materialized.

__global__ __launch_bounds__(256) void slot_attention(
    const float* __restrict__ noise, const float* __restrict__ mu,
    const float* __restrict__ sigma,
    const float* __restrict__ ln_slot_g, const float* __restrict__ ln_slot_b,
    const float* __restrict__ ln_mlp_g,  const float* __restrict__ ln_mlp_b,
    const float* __restrict__ Wq,
    const float* __restrict__ W_ih, const float* __restrict__ W_hh,
    const float* __restrict__ b_ih, const float* __restrict__ b_hh,
    const float* __restrict__ mlp_W1, const float* __restrict__ mlp_b1,
    const float* __restrict__ mlp_W2, const float* __restrict__ mlp_b2,
    const bf16_t* __restrict__ kb, const bf16_t* __restrict__ vb,
    float* __restrict__ out)
{
    __shared__ bf16_t s_k[8][2][32][64];   // 64 KB, per-wave double buffers
    __shared__ bf16_t s_v[8][2][32][64];   // 64 KB
    __shared__ bf16_t s_q[16][64];         // padded q, shared read-only
    __shared__ bf16_t s_a[8][16][32];      // per-wave softmax weights
    __shared__ float  s_slots[6][64];
    __shared__ float  s_ln[6][64];
    __shared__ float  s_upd[6][64];        // U accumulator (ds_add_f32)
    __shared__ float  s_z[6];              // Z accumulator (ds_add_f32)
    __shared__ float  s_gi[6][192];
    __shared__ float  s_gh[6][192];
    __shared__ float  s_hid[6][128];

    const int tid  = threadIdx.x;
    const int lane = tid & 31;
    const int wid  = tid >> 5;
    const int b = blockIdx.x;
    const bf16_t* kbB = kb + (long)b * 4096 * 64;
    const bf16_t* vbB = vb + (long)b * 4096 * 64;

    // slots = mu + sigma * noise; zero padded q/a rows once
    for (int i = tid; i < 6 * 64; i += 256) {
        const int h = i & 63;
        (&s_slots[0][0])[i] = mu[h] + sigma[h] * noise[b * 384 + i];
    }
    for (int i = tid; i < 16 * 64; i += 256) (&s_q[0][0])[i] = (bf16_t)0.f;
    for (int i = tid; i < 8 * 16 * 32; i += 256) (&s_a[0][0][0])[i] = (bf16_t)0.f;
    __syncthreads();

    const float scale = 0.125f;  // 64^-0.5

    for (int it = 0; it < 3; ++it) {
        // --- LN(slots) by wave 0 ---
        if (wid == 0) {
            for (int s = 0; s < 6; ++s) {
                const float a0 = s_slots[s][lane], a1 = s_slots[s][lane + 32];
                float sm = a0 + a1;
                for (int m = 16; m > 0; m >>= 1) sm += __shfl_xor(sm, m, 32);
                const float mean = sm * (1.f / 64.f);
                const float d0 = a0 - mean, d1 = a1 - mean;
                float vs = d0 * d0 + d1 * d1;
                for (int m = 16; m > 0; m >>= 1) vs += __shfl_xor(vs, m, 32);
                const float rstd = rsqrtf(vs * (1.f / 64.f) + 1e-5f);
                s_ln[s][lane]      = d0 * rstd * ln_slot_g[lane] + ln_slot_b[lane];
                s_ln[s][lane + 32] = d1 * rstd * ln_slot_g[lane + 32] + ln_slot_b[lane + 32];
            }
        }
        __syncthreads();

        // --- q = LN(slots) @ Wq^T (bf16), zero U/Z accumulators ---
        for (int i = tid; i < 6 * 64; i += 256) {
            const int s = i >> 6, g = i & 63;
            const float* wr = Wq + g * 64;
            float acc = 0.f;
            for (int h2 = 0; h2 < 64; ++h2) acc += s_ln[s][h2] * wr[h2];
            s_q[s][g] = (bf16_t)acc;
            (&s_upd[0][0])[i] = 0.f;
        }
        if (tid < 6) s_z[tid] = 0.f;
        __syncthreads();

        const v16bf qf0 = frag_a_16x32(&s_q[lane & 15][0]);
        const v16bf qf1 = frag_a_16x32(&s_q[lane & 15][32]);

        v8f cU[4];
        const v8f vzero = {0.f, 0.f, 0.f, 0.f, 0.f, 0.f, 0.f, 0.f};
#pragma unroll
        for (int g = 0; g < 4; ++g) cU[g] = vzero;
        float zacc[6] = {0.f, 0.f, 0.f, 0.f, 0.f, 0.f};

        // --- barrier-free streaming over this wave's 16 chunks (512 points) ---
        const int nc0 = wid * 16;
        stage_chunk(&s_k[wid][0][0][0], &s_v[wid][0][0][0],
                    kbB + (long)nc0 * 2048, vbB + (long)nc0 * 2048, lane);
        for (int c = 0; c < 16; ++c) {
            const int cur = c & 1;
            if (c + 1 < 16)
                stage_chunk(&s_k[wid][cur ^ 1][0][0], &s_v[wid][cur ^ 1][0][0],
                            kbB + (long)(nc0 + c + 1) * 2048,
                            vbB + (long)(nc0 + c + 1) * 2048, lane);
            wait_chunk(c + 1 < 16);

            // logits tiles [16 slots x 16 points], K = 64 = 2 * 32
#pragma unroll
            for (int nt = 0; nt < 2; ++nt) {
                v8f cl = vzero;
                cl = wmma_bf16(qf0, frag_b_contig(&s_k[wid][cur][nt * 16 + (lane & 15)][0]),  cl);
                cl = wmma_bf16(qf1, frag_b_contig(&s_k[wid][cur][nt * 16 + (lane & 15)][32]), cl);
                // column n's 6 valid slot logits live in lane n (rows 0..5)
                if (lane < 16) {
                    float l[6];
                    float mx = -1e30f;
#pragma unroll
                    for (int s = 0; s < 6; ++s) { l[s] = cl[s] * scale; mx = fmaxf(mx, l[s]); }
                    float sum = 0.f;
#pragma unroll
                    for (int s = 0; s < 6; ++s) { l[s] = __expf(l[s] - mx); sum += l[s]; }
                    const float inv = 1.f / sum;
                    const int ncol = nt * 16 + lane;
#pragma unroll
                    for (int s = 0; s < 6; ++s) {
                        const float a = l[s] * inv;
                        zacc[s] += a;
                        s_a[wid][s][ncol] = (bf16_t)a;   // same-wave DS ops stay in order
                    }
                }
            }

            // U += a[16 x 32] @ v[32 x 64]
            const v16bf af = frag_a_16x32(&s_a[wid][lane & 15][0]);
#pragma unroll
            for (int g = 0; g < 4; ++g)
                cU[g] = wmma_bf16(af,
                                  frag_b_strided(&s_v[wid][cur][0][g * 16 + (lane & 15)], 64),
                                  cU[g]);
        }

        // --- cross-wave merge: ds_add_f32 atomics ---
        if (lane < 16) {
#pragma unroll
            for (int s = 0; s < 6; ++s) atomicAdd(&s_z[s], zacc[s]);
#pragma unroll
            for (int g = 0; g < 4; ++g)
#pragma unroll
                for (int r = 0; r < 6; ++r)
                    atomicAdd(&s_upd[r][g * 16 + lane], cU[g][r]);
        }
        __syncthreads();

        // updates = U / (Z + 1e-8)
        for (int i = tid; i < 6 * 64; i += 256) {
            const int s = i >> 6;
            (&s_upd[0][0])[i] = (&s_upd[0][0])[i] / (s_z[s] + 1e-8f);
        }
        __syncthreads();

        // --- GRUCell: x = updates, h = slots ---
        for (int i = tid; i < 6 * 192; i += 256) {
            const int s = i / 192, j = i % 192;
            const float* wi = W_ih + j * 64;
            const float* wh = W_hh + j * 64;
            float ai = b_ih[j], ah = b_hh[j];
            for (int h2 = 0; h2 < 64; ++h2) {
                ai += wi[h2] * s_upd[s][h2];
                ah += wh[h2] * s_slots[s][h2];
            }
            s_gi[s][j] = ai;
            s_gh[s][j] = ah;
        }
        __syncthreads();
        for (int i = tid; i < 6 * 64; i += 256) {
            const int s = i >> 6, h = i & 63;
            const float r = sigm(s_gi[s][h] + s_gh[s][h]);
            const float z = sigm(s_gi[s][64 + h] + s_gh[s][64 + h]);
            const float n = tanhf(s_gi[s][128 + h] + r * s_gh[s][128 + h]);
            (&s_slots[0][0])[i] = (1.f - z) * n + z * s_slots[s][h];
        }
        __syncthreads();

        // --- residual MLP ---
        if (wid == 0) {
            for (int s = 0; s < 6; ++s) {
                const float a0 = s_slots[s][lane], a1 = s_slots[s][lane + 32];
                float sm = a0 + a1;
                for (int m = 16; m > 0; m >>= 1) sm += __shfl_xor(sm, m, 32);
                const float mean = sm * (1.f / 64.f);
                const float d0 = a0 - mean, d1 = a1 - mean;
                float vs = d0 * d0 + d1 * d1;
                for (int m = 16; m > 0; m >>= 1) vs += __shfl_xor(vs, m, 32);
                const float rstd = rsqrtf(vs * (1.f / 64.f) + 1e-5f);
                s_ln[s][lane]      = d0 * rstd * ln_mlp_g[lane] + ln_mlp_b[lane];
                s_ln[s][lane + 32] = d1 * rstd * ln_mlp_g[lane + 32] + ln_mlp_b[lane + 32];
            }
        }
        __syncthreads();
        for (int i = tid; i < 6 * 128; i += 256) {
            const int s = i >> 7, j = i & 127;
            const float* w1 = mlp_W1 + j * 64;
            float acc = mlp_b1[j];
            for (int h2 = 0; h2 < 64; ++h2) acc += w1[h2] * s_ln[s][h2];
            s_hid[s][j] = fmaxf(acc, 0.f);
        }
        __syncthreads();
        for (int i = tid; i < 6 * 64; i += 256) {
            const int s = i >> 6, h = i & 63;
            const float* w2 = mlp_W2 + h * 128;
            float acc = mlp_b2[h];
            for (int j = 0; j < 128; ++j) acc += w2[j] * s_hid[s][j];
            (&s_slots[0][0])[i] += acc;
        }
        __syncthreads();
    }

    for (int i = tid; i < 6 * 64; i += 256)
        out[b * 384 + i] = (&s_slots[0][0])[i];
}

// ---- Launch -----------------------------------------------------------------

extern "C" void kernel_launch(void* const* d_in, const int* in_sizes, int n_in,
                              void* d_out, int out_size, void* d_ws, size_t ws_size,
                              hipStream_t stream) {
    const float* features   = (const float*)d_in[0];
    const float* slot_noise = (const float*)d_in[1];
    const float* slot_mu    = (const float*)d_in[2];
    const float* slot_sigma = (const float*)d_in[3];
    const float* ln_feat_g  = (const float*)d_in[4];
    const float* ln_feat_b  = (const float*)d_in[5];
    const float* ln_slot_g  = (const float*)d_in[6];
    const float* ln_slot_b  = (const float*)d_in[7];
    const float* ln_mlp_g   = (const float*)d_in[8];
    const float* ln_mlp_b   = (const float*)d_in[9];
    const float* Wk         = (const float*)d_in[10];
    const float* Wv         = (const float*)d_in[11];
    const float* Wq         = (const float*)d_in[12];
    const float* W_ih       = (const float*)d_in[13];
    const float* W_hh       = (const float*)d_in[14];
    const float* b_ih       = (const float*)d_in[15];
    const float* b_hh       = (const float*)d_in[16];
    const float* mlp_W1     = (const float*)d_in[17];
    const float* mlp_b1     = (const float*)d_in[18];
    const float* mlp_W2     = (const float*)d_in[19];
    const float* mlp_b2     = (const float*)d_in[20];

    const size_t kv_elems = (size_t)64 * 4096 * 64;
    bf16_t* kb  = (bf16_t*)d_ws;
    bf16_t* vb  = kb + kv_elems;
    bf16_t* wkb = vb + kv_elems;
    bf16_t* wvb = wkb + 64 * 256;

    convert_weights<<<64, 256, 0, stream>>>(Wk, Wv, wkb, wvb);
    ln_project<<<16384, 32, 0, stream>>>(features, ln_feat_g, ln_feat_b,
                                         wkb, wvb, kb, vb);
    slot_attention<<<64, 256, 0, stream>>>(slot_noise, slot_mu, slot_sigma,
                                           ln_slot_g, ln_slot_b, ln_mlp_g, ln_mlp_b,
                                           Wq, W_ih, W_hh, b_ih, b_hh,
                                           mlp_W1, mlp_b1, mlp_W2, mlp_b2,
                                           kb, vb, (float*)d_out);
}
